// FullyNonlinearKANCell_53283364274888
// MI455X (gfx1250) — matrix operand order repaired and checked by reading, not compiled, not run on hardware
//
#include <hip/hip_runtime.h>
#include <hip/hip_bf16.h>

#define B_    4096
#define IN_   256
#define HID_  512
#define KB_   16
#define KD_   (IN_ * KB_)   // 4096 contraction length

typedef __attribute__((ext_vector_type(16))) _Float16 v16h;
typedef __attribute__((ext_vector_type(8)))  _Float16 v8h;
typedef __attribute__((ext_vector_type(4)))  _Float16 v4h;
typedef __attribute__((ext_vector_type(8)))  float    v8f;
typedef __attribute__((ext_vector_type(4)))  float    v4f;

__device__ __forceinline__ float fast_tanh(float a) {
#if __has_builtin(__builtin_amdgcn_tanhf)
    return __builtin_amdgcn_tanhf(a);
#else
    float e = __expf(-2.0f * a);
    return (1.0f - e) / (1.0f + e);
#endif
}

__device__ __forceinline__ float fast_sigmoid(float a) {
    return 1.0f / (1.0f + __expf(-a));
}

// ---------------------------------------------------------------------------
// Kernel 1: phi[b, i*16+k] = tanh((x[b,i]-c[k])*s[k]), stored f16, row-major
// (B, KD). One thread per (b,i): 16 tanh, two 16-byte stores. Regular (RT)
// stores so the 32 MB phi tensor stays resident in the 192 MB L2 for the GEMM.
// ---------------------------------------------------------------------------
__global__ void phi_kernel(const float* __restrict__ x,
                           const float* __restrict__ c,
                           const float* __restrict__ s,
                           _Float16* __restrict__ phi) {
    int idx = blockIdx.x * blockDim.x + threadIdx.x;   // b*IN + i
    if (idx >= B_ * IN_) return;
    float xv = x[idx];
    _Float16 p[KB_];
#pragma unroll
    for (int k = 0; k < KB_; ++k) {
        p[k] = (_Float16)fast_tanh((xv - c[k]) * s[k]);
    }
    v8h* dst = (v8h*)(phi + (size_t)idx * KB_);
    dst[0] = *(v8h*)&p[0];
    dst[1] = *(v8h*)&p[8];
}

// ---------------------------------------------------------------------------
// Kernel 2: Wx f32 -> f16. Wx is (HID, IN, K) row-major == (HID, KD) row-major,
// which is exactly the B-operand layout (column n of B = row n of w16,
// contiguous along the contraction).
// ---------------------------------------------------------------------------
__global__ void wcvt_kernel(const float* __restrict__ w,
                            _Float16* __restrict__ w16, int n4) {
    int idx = blockIdx.x * blockDim.x + threadIdx.x;
    if (idx >= n4) return;
    v4f v = ((const v4f*)w)[idx];
    v4h h = {(_Float16)v[0], (_Float16)v[1], (_Float16)v[2], (_Float16)v[3]};
    ((v4h*)w16)[idx] = h;
}

// ---------------------------------------------------------------------------
// Kernel 3: WMMA GEMM  out = sigmoid(phi @ w16^T + bias)
//   phi : (B, KD) f16 row-major      (A operand)
//   w16 : (HID, KD) f16 row-major    (B operand)
//   out : (B, HID) f32
//
// Workgroup = 256 threads = 8 waves. M-block = 32 rows shared by all waves
// (WGP$ dedups the identical A-frag loads); each wave owns a 64-col N-slice.
// Wave tile = 32(M) x 64(N) = 2x4 WMMA tiles -> 8 independent v_wmma per
// k-step from 2 A-frags + 4 B-frags (1.5 b128 loads per WMMA).
// Workgroup covers the FULL N=512 extent, so each phi row is streamed once.
// Grid = (4096/32, 1) = 128 workgroups = 1024 waves.
//
// Fragment lane mapping (wave32, 16-bit, per ISA 7.12.2):
//   A 16x32 : lane L (r=L&15, hi=L>>4) row=r, K in {hi*8..+8} u {16+hi*8..+8}
//   B 32x16 : lane L col=r, K in {hi*8..+8} u {16+hi*8..+8}  (same as A)
//   C/D     : VGPR v, lane L -> row = v + hi*8, col = r
// ---------------------------------------------------------------------------
__global__ void __launch_bounds__(256)
gemm_kernel(const _Float16* __restrict__ phi,
            const _Float16* __restrict__ w16,
            const float* __restrict__ bias,
            float* __restrict__ out) {
    const int lane = threadIdx.x & 31;
    const int wave = threadIdx.x >> 5;
    const int r    = lane & 15;
    const int hi   = lane >> 4;

    const int m0 = blockIdx.x * 32;   // 32 rows per workgroup (shared by waves)
    const int n0 = wave * 64;         // 64 cols per wave

    v8f acc[2][4] = {};

    const _Float16* aRow0 = phi + (size_t)(m0 + r) * KD_ + hi * 8;
    const _Float16* aRow1 = aRow0 + (size_t)16 * KD_;
    const _Float16* bRow[4];
#pragma unroll
    for (int j = 0; j < 4; ++j)
        bRow[j] = w16 + (size_t)(n0 + j * 16 + r) * KD_ + hi * 8;

#pragma unroll 2
    for (int k0 = 0; k0 < KD_; k0 += 32) {
        v8h a0lo = *(const v8h*)(aRow0 + k0);
        v8h a0hi = *(const v8h*)(aRow0 + k0 + 16);
        v8h a1lo = *(const v8h*)(aRow1 + k0);
        v8h a1hi = *(const v8h*)(aRow1 + k0 + 16);
        v16h A0 = __builtin_shufflevector(a0lo, a0hi,
                    0,1,2,3,4,5,6,7,8,9,10,11,12,13,14,15);
        v16h A1 = __builtin_shufflevector(a1lo, a1hi,
                    0,1,2,3,4,5,6,7,8,9,10,11,12,13,14,15);

        v16h Bf[4];
#pragma unroll
        for (int j = 0; j < 4; ++j)
            Bf[j] = __builtin_shufflevector(*(const v8h*)(bRow[j] + k0), *(const v8h*)(bRow[j] + k0 + 16),
                    0,1,2,3,4,5,6,7,8,9,10,11,12,13,14,15);

#pragma unroll
        for (int j = 0; j < 4; ++j) {
            acc[0][j] = __builtin_amdgcn_wmma_f32_16x16x32_f16(
                false, A0, false, Bf[j], (short)0, acc[0][j], false, false);
            acc[1][j] = __builtin_amdgcn_wmma_f32_16x16x32_f16(
                false, A1, false, Bf[j], (short)0, acc[1][j], false, false);
        }
    }

#pragma unroll
    for (int j = 0; j < 4; ++j) {
        const int col = n0 + j * 16 + r;
        const float bb = bias[col];
#pragma unroll
        for (int v = 0; v < 8; ++v) {
            const int row0 = m0 + hi * 8 + v;
            out[(size_t)row0 * HID_ + col]        = fast_sigmoid(acc[0][j][v] + bb);
            out[(size_t)(row0 + 16) * HID_ + col] = fast_sigmoid(acc[1][j][v] + bb);
        }
    }
}

// ---------------------------------------------------------------------------
// Launch. Inputs (setup_inputs order):
//   0:x_t 1:h_prev 2:Wx 3:cx 4:sx 5:bx 6:Wh 7:ch 8:sh 9:bh
// The slice sigmoid(concat)[:, :HID] makes the whole h-path dead code, so
// only inputs 0,2,3,4,5 are read.
// Workspace: phi (B*KD f16 = 32 MB) | w16 (HID*KD f16 = 4 MB) = 36 MB.
// ---------------------------------------------------------------------------
extern "C" void kernel_launch(void* const* d_in, const int* in_sizes, int n_in,
                              void* d_out, int out_size, void* d_ws, size_t ws_size,
                              hipStream_t stream) {
    const float* x_t = (const float*)d_in[0];
    const float* Wx  = (const float*)d_in[2];
    const float* cx  = (const float*)d_in[3];
    const float* sx  = (const float*)d_in[4];
    const float* bx  = (const float*)d_in[5];

    _Float16* phi = (_Float16*)d_ws;
    _Float16* w16 = (_Float16*)((char*)d_ws + (size_t)B_ * KD_ * sizeof(_Float16));

    {
        int n = B_ * IN_;
        phi_kernel<<<(n + 255) / 256, 256, 0, stream>>>(x_t, cx, sx, phi);
    }
    {
        int n4 = (HID_ * KD_) / 4;
        wcvt_kernel<<<(n4 + 255) / 256, 256, 0, stream>>>(Wx, w16, n4);
    }
    {
        dim3 grid(B_ / 32, 1);
        gemm_kernel<<<grid, 256, 0, stream>>>(phi, w16, bx, (float*)d_out);
    }
}
